// NTXentLoss_11141145166516
// MI455X (gfx1250) — matrix operand order repaired and checked by reading, not compiled, not run on hardware
//
#include <hip/hip_runtime.h>

typedef __attribute__((ext_vector_type(16))) _Float16 v16h;
typedef __attribute__((ext_vector_type(8)))  _Float16 v8h;
typedef __attribute__((ext_vector_type(4)))  _Float16 v4h;
typedef __attribute__((ext_vector_type(8)))  float    v8f;
typedef __attribute__((ext_vector_type(4)))  float    v4f;
typedef __attribute__((ext_vector_type(4)))  unsigned int v4u;
typedef __attribute__((ext_vector_type(4)))  int      v4i;
typedef __attribute__((ext_vector_type(8)))  int      v8i;

#define NROWS   8192           // 2N
#define NHALF   4096           // N
#define D       128
#define TINV    2.0f           // 1 / TEMPERATURE
#define NTILE   512            // NROWS / 16

#define CH_TILES 16                    // column tiles per LDS chunk
#define CH_ROWS  (CH_TILES * 16)       // 256 Zn rows per chunk
#define NCHUNK   (NTILE / CH_TILES)    // 32 chunks
#define ROWPAD_H 136                   // 128 halves + 16B pad -> 272B LDS row stride
#define BUF_H    (CH_ROWS * ROWPAD_H)  // halves per LDS buffer
#define SMEM_BYTES (2u * BUF_H * sizeof(_Float16))   // 136 KB (double buffer)

#if __has_builtin(__builtin_amdgcn_tensor_load_to_lds) && \
    __has_builtin(__builtin_amdgcn_s_wait_tensorcnt)
#define USE_TDM 1
#else
#define USE_TDM 0
#endif

// ---------------------------------------------------------------------------
// Kernel 1: row L2-normalize. One wave per row; writes f16 Zn + clamped norms.
// ---------------------------------------------------------------------------
__global__ __launch_bounds__(256)
void ntx_normalize(const float* __restrict__ z_i, const float* __restrict__ z_j,
                   _Float16* __restrict__ Zh, float* __restrict__ nrm) {
  int wid  = (blockIdx.x * blockDim.x + threadIdx.x) >> 5;
  int lane = threadIdx.x & 31;
  if (wid >= NROWS) return;
  const float* src = (wid < NHALF) ? (z_i + (size_t)wid * D)
                                   : (z_j + (size_t)(wid - NHALF) * D);
  v4f v = *(const v4f*)(src + lane * 4);
  float ss = v[0]*v[0] + v[1]*v[1] + v[2]*v[2] + v[3]*v[3];
  #pragma unroll
  for (int m = 16; m >= 1; m >>= 1) ss += __shfl_xor(ss, m, 32);
  float nn = fmaxf(sqrtf(ss), 1e-8f);
  float sc = 1.0f / nn;
  if (lane == 0) nrm[wid] = nn;
  v4h o;
  o[0] = (_Float16)(v[0] * sc);
  o[1] = (_Float16)(v[1] * sc);
  o[2] = (_Float16)(v[2] * sc);
  o[3] = (_Float16)(v[3] * sc);
  *(v4h*)(Zh + (size_t)wid * D + lane * 4) = o;
}

// ---------------------------------------------------------------------------
// Kernel 2: fused sim + streaming softmax-CE.
// Block = 4 waves; wave wv owns row-block rt = blockIdx*4+wv. All 4 waves
// consume shared B chunks (256 Zn rows) staged in double-buffered LDS by the
// Tensor Data Mover (TDM pads rows to 272B for bank-conflict-free b128 reads).
// B fragments are software-pipelined across tiles so ds_load latency overlaps
// the previous tile's WMMA + exp work.
// ---------------------------------------------------------------------------
__global__ __launch_bounds__(128)
void ntx_simloss(const float* __restrict__ z_i, const float* __restrict__ z_j,
                 const _Float16* __restrict__ Zh, const float* __restrict__ nrm,
                 float* __restrict__ partials) {
  extern __shared__ _Float16 smem[];   // 2 x BUF_H halves

  int lane = threadIdx.x & 31;
  int wv   = threadIdx.x >> 5;          // 0..3
  int rt   = blockIdx.x * 4 + wv;       // row-block 0..511
  int m    = lane & 15;
  int hi   = lane >> 4;

  // ---- A fragments (16 rows x K=128), f16 16x32 WMMA layout, kept in regs
  v16h a[4];
  const _Float16* arow = Zh + (size_t)(rt * 16 + m) * D;
  #pragma unroll
  for (int kb = 0; kb < 4; ++kb) {
    const _Float16* p = arow + kb * 32 + hi * 8;
    v8h lo = *(const v8h*)p;
    v8h hh = *(const v8h*)(p + 16);
    #pragma unroll
    for (int i = 0; i < 8; ++i) { a[kb][i] = lo[i]; a[kb][i + 8] = hh[i]; }
  }

  float s[8] = {0.f, 0.f, 0.f, 0.f, 0.f, 0.f, 0.f, 0.f};

#if USE_TDM
  // ---- TDM D# per ISA 8.3-8.6: 2-D tile 128x256 f16 out of the 8192x128
  // tensor, LDS padding 4 DWORDs every 64 DWORDs (272B row stride).
  auto issue_chunk = [&](int ch) {
    unsigned parity  = (unsigned)(ch & 1);
    unsigned lds_off = (unsigned)(uintptr_t)(const void*)smem
                     + parity * (unsigned)(BUF_H * sizeof(_Float16));
    unsigned long long ga =
        (unsigned long long)(const void*)(Zh + (size_t)ch * CH_ROWS * D);
    v4u g0;
    g0[0] = 1u;                                   // count=1, user mode
    g0[1] = lds_off;                              // lds_addr
    g0[2] = (unsigned)ga;                         // global_addr[31:0]
    g0[3] = ((unsigned)(ga >> 32) & 0x01FFFFFFu)  // global_addr[56:32]
          | (2u << 30);                           // type=2 (image)
    v8i g1;
    g1[0] = (1 << 16)      // data_size = 1 -> 2 bytes
          | (1 << 20)      // pad_enable
          | (5 << 22)      // pad_interval: every 64 DWORDs (one 128-half row)
          | (3 << 25);     // pad_amount: 4 DWORDs (16B)
    g1[1] = (int)((unsigned)D << 16);                 // tensor_dim0 = 128
    g1[2] = (int)((unsigned)(NROWS & 0xFFFF) << 16);  // tensor_dim1 = 8192
    g1[3] = (int)((unsigned)D << 16);                 // tile_dim0 = 128
    g1[4] = CH_ROWS;                                  // tile_dim1 = 256
    g1[5] = D;                                        // tensor_dim0_stride
    g1[6] = 0;
    g1[7] = 0;
    v4i gz;  gz[0] = gz[1] = gz[2] = gz[3] = 0;       // D# groups 2/3: unused
    v8i gz8; gz8[0] = gz8[1] = gz8[2] = gz8[3] = 0;
             gz8[4] = gz8[5] = gz8[6] = gz8[7] = 0;
    __builtin_amdgcn_tensor_load_to_lds(g0, g1, gz, gz, gz8, 0);
  };
  if (wv == 0) {
    issue_chunk(0);
    __builtin_amdgcn_s_wait_tensorcnt(0);
  }
#else
  // ---- fallback: cooperative global->LDS copy into the same padded layout
  auto issue_chunk = [&](int ch) {
    _Float16* dst = smem + (size_t)(ch & 1) * BUF_H;
    const _Float16* src = Zh + (size_t)ch * CH_ROWS * D;
    #pragma unroll
    for (int rep = 0; rep < 2; ++rep) {
      int row = rep * 128 + (int)threadIdx.x;
      const _Float16* s0 = src + (size_t)row * D;
      _Float16* d0 = dst + (size_t)row * ROWPAD_H;
      #pragma unroll
      for (int k = 0; k < 16; ++k)
        *(v8h*)(d0 + k * 8) = *(const v8h*)(s0 + k * 8);
    }
  };
  issue_chunk(0);
#endif
  __syncthreads();

  for (int ch = 0; ch < NCHUNK; ++ch) {
#if USE_TDM
    if (wv == 0 && ch + 1 < NCHUNK) issue_chunk(ch + 1);   // DMA next chunk
#else
    if (ch + 1 < NCHUNK) issue_chunk(ch + 1);
#endif

    // lane's base address inside the chunk: column n=m, K = kb*32 + hi*8 (+16 for upper 8)
    const _Float16* bln = smem + (size_t)(ch & 1) * BUF_H
                        + (size_t)m * ROWPAD_H + hi * 8;

    // preload tile 0's B fragments
    v16h bcur[4];
    #pragma unroll
    for (int kb = 0; kb < 4; ++kb) {
      v8h l0 = *(const v8h*)(bln + kb * 32);
      v8h l1 = *(const v8h*)(bln + kb * 32 + 16);
      #pragma unroll
      for (int i = 0; i < 8; ++i) { bcur[kb][i] = l0[i]; bcur[kb][i + 8] = l1[i]; }
    }

    #pragma unroll 4
    for (int t = 0; t < CH_TILES; ++t) {
      // issue next tile's 8 ds_load_b128 before consuming the current tile
      v16h bnxt[4];
      if (t + 1 < CH_TILES) {
        const _Float16* bn = bln + (size_t)((t + 1) * 16) * ROWPAD_H;
        #pragma unroll
        for (int kb = 0; kb < 4; ++kb) {
          v8h l0 = *(const v8h*)(bn + kb * 32);
          v8h l1 = *(const v8h*)(bn + kb * 32 + 16);
          #pragma unroll
          for (int i = 0; i < 8; ++i) { bnxt[kb][i] = l0[i]; bnxt[kb][i + 8] = l1[i]; }
        }
      }

      int jt = ch * CH_TILES + t;
      v8f c = {0.f, 0.f, 0.f, 0.f, 0.f, 0.f, 0.f, 0.f};
      #pragma unroll
      for (int kb = 0; kb < 4; ++kb) {
        c = __builtin_amdgcn_wmma_f32_16x16x32_f16(
                false, a[kb], false, bcur[kb], (short)0, c, false, false);
      }
      bool dt = (jt == rt);
      #pragma unroll
      for (int r = 0; r < 8; ++r) {
        float x = c[r] * TINV;
        if (dt && (lane == r || lane == r + 24)) x = -__builtin_inff();
        s[r] += __expf(x);
      }

      if (t + 1 < CH_TILES) {
        #pragma unroll
        for (int kb = 0; kb < 4; ++kb) bcur[kb] = bnxt[kb];
      }
    }

#if USE_TDM
    if (wv == 0) __builtin_amdgcn_s_wait_tensorcnt(0);     // chunk ch+1 landed
#endif
    __syncthreads();   // all waves done with buf(ch) before it is overwritten
  }

  // reduce sum(exp) across the 16 lanes of each half
  #pragma unroll
  for (int r = 0; r < 8; ++r) {
    float t = s[r];
    t += __shfl_xor(t, 1, 32);
    t += __shfl_xor(t, 2, 32);
    t += __shfl_xor(t, 4, 32);
    t += __shfl_xor(t, 8, 32);
    s[r] = t;
  }

  // exact fp32 positive-pair similarity; lanes (m, m+16) split the 128-dim dot
  int ra = rt * 16 + m;
  int rb = (ra < NHALF) ? ra + NHALF : ra - NHALF;
  const float* pa = (ra < NHALF) ? z_i + (size_t)ra * D
                                 : z_j + (size_t)(ra - NHALF) * D;
  const float* pb = (rb < NHALF) ? z_i + (size_t)rb * D
                                 : z_j + (size_t)(rb - NHALF) * D;
  float part = 0.f;
  #pragma unroll
  for (int k = 0; k < 16; ++k) {
    v4f xa = *(const v4f*)(pa + hi * 64 + k * 4);
    v4f xb = *(const v4f*)(pb + hi * 64 + k * 4);
    part += xa[0]*xb[0] + xa[1]*xb[1] + xa[2]*xb[2] + xa[3]*xb[3];
  }
  part += __shfl_xor(part, 16, 32);
  float posdot = part / (nrm[ra] * nrm[rb]);

  // per-row contribution with static register indexing:
  // lane r (r<8) owns row r (s[r], pos m=r); lane 24+r owns row 8+r.
  float contrib = 0.f;
  #pragma unroll
  for (int r = 0; r < 8; ++r) {
    if (lane == r || lane == r + 24)
      contrib = __logf(s[r]) - posdot * TINV;
  }
  #pragma unroll
  for (int mm = 16; mm >= 1; mm >>= 1) contrib += __shfl_xor(contrib, mm, 32);
  if (lane == 0) partials[rt] = contrib;
}

// ---------------------------------------------------------------------------
// Kernel 3: deterministic fixed-order reduction of 512 partials -> mean loss.
// ---------------------------------------------------------------------------
__global__ __launch_bounds__(32)
void ntx_finalize(const float* __restrict__ partials, float* __restrict__ out) {
  int lane = threadIdx.x;
  float s = 0.f;
  #pragma unroll
  for (int i = 0; i < 16; ++i) s += partials[lane * 16 + i];
  #pragma unroll
  for (int mm = 16; mm >= 1; mm >>= 1) s += __shfl_xor(s, mm, 32);
  if (lane == 0) out[0] = s * (1.0f / (float)NROWS);
}

extern "C" void kernel_launch(void* const* d_in, const int* in_sizes, int n_in,
                              void* d_out, int out_size, void* d_ws, size_t ws_size,
                              hipStream_t stream) {
  (void)in_sizes; (void)n_in; (void)out_size; (void)ws_size;
  const float* z_i = (const float*)d_in[0];
  const float* z_j = (const float*)d_in[1];
  float* out = (float*)d_out;

  _Float16* Zh = (_Float16*)d_ws;                                // 2 MB
  float* nrm   = (float*)((char*)d_ws + (size_t)NROWS * D * 2);  // 32 KB
  float* parts = nrm + NROWS;                                    // 512 floats

  ntx_normalize<<<1024, 256, 0, stream>>>(z_i, z_j, Zh, nrm);
  ntx_simloss<<<128, 128, SMEM_BYTES, stream>>>(z_i, z_j, Zh, nrm, parts);
  ntx_finalize<<<1, 32, 0, stream>>>(parts, out);
}
